// Sequential2D_55276229099866
// MI455X (gfx1250) — matrix operand order repaired and checked by reading, not compiled, not run on hardware
//
#include <hip/hip_runtime.h>

typedef __attribute__((ext_vector_type(16))) __bf16 v16bf;
typedef __attribute__((ext_vector_type(8)))  float  v8f;

#define N_OUT  8
#define N_IN   8
#define DDIM   128
#define BATCH  32768
#define BAND   2
#define TILE_B 128
#define SH     136   // LDS row stride in 16-bit units (272B: 4-bank skew, 16B-aligned rows)

union FragU { v16bf v; uint4 u[2]; };
union Pack4 { unsigned short h[4]; uint2 u2; };

__device__ __forceinline__ unsigned short f2bf_rne(float x) {
    unsigned u = __float_as_uint(x);
    unsigned r = ((u >> 16) & 1u) + 0x7FFFu;   // round-to-nearest-even
    return (unsigned short)((u + r) >> 16);
}
__device__ __forceinline__ float bf2f(unsigned short h) {
    return __uint_as_float(((unsigned)h) << 16);
}

// ---- prep: split W (8*8*128*128 f32) into bf16 hi/lo planes in workspace ----
__global__ void seq2d_prep_w(const float* __restrict__ W,
                             unsigned short* __restrict__ Wh,
                             unsigned short* __restrict__ Wl) {
    const size_t idx = ((size_t)blockIdx.x * 256 + threadIdx.x) * 4;
    const float4 v = *(const float4*)(W + idx);
    float xs[4] = {v.x, v.y, v.z, v.w};
    Pack4 ph, pl;
#pragma unroll
    for (int k = 0; k < 4; ++k) {
        unsigned short hb = f2bf_rne(xs[k]);
        ph.h[k] = hb;
        pl.h[k] = f2bf_rne(xs[k] - bf2f(hb));
    }
    *(uint2*)(Wh + idx) = ph.u2;
    *(uint2*)(Wl + idx) = pl.u2;
}

// ---- prep: biasSum[i][o] = sum over active j of b[i][j][o] ----
__global__ void seq2d_prep_bias(const float* __restrict__ b, float* __restrict__ bs) {
    const int t = blockIdx.x * 256 + threadIdx.x;   // 1024 total
    const int i = t >> 7, o = t & 127;
    int j0 = i - BAND; if (j0 < 0) j0 = 0;
    float s = 0.f;
    for (int j = j0; j <= i; ++j) s += b[(i * N_IN + j) * DDIM + o];
    bs[t] = s;
}

// ---- main: 128(batch) x 128(out) tile per workgroup, bf16x3 WMMA ----
__global__ __launch_bounds__(256)
void seq2d_main(const float* __restrict__ X,
                const unsigned short* __restrict__ Wh,
                const unsigned short* __restrict__ Wl,
                const float* __restrict__ biasSum,
                float* __restrict__ out) {
    __shared__ __align__(16) unsigned short smem[4 * TILE_B * SH];
    unsigned short* XhL = smem;
    unsigned short* XlL = smem + TILE_B * SH;
    unsigned short* WhL = smem + 2 * TILE_B * SH;
    unsigned short* WlL = smem + 3 * TILE_B * SH;

    const int i   = blockIdx.y;
    const int b0  = blockIdx.x * TILE_B;
    const int tid = threadIdx.x;
    const int ln  = tid & 31;
    const int wv  = tid >> 5;
    const int m0  = (wv & 3) * 32;   // wave's M (batch) base: 2 tiles of 16
    const int n0  = (wv >> 2) * 64;  // wave's N (out) base: 4 tiles of 16
    const int lr  = ln & 15;
    const int g   = ln >> 4;

    v8f acc[2][4];
#pragma unroll
    for (int mt = 0; mt < 2; ++mt)
#pragma unroll
        for (int nt = 0; nt < 4; ++nt)
#pragma unroll
            for (int q = 0; q < 8; ++q) acc[mt][nt][q] = 0.0f;

    int j0 = i - BAND; if (j0 < 0) j0 = 0;

    for (int j = j0; j <= i; ++j) {
        __syncthreads();
        // stage X[j] tile: f32 -> bf16 hi/lo into LDS (coalesced float4 loads)
        {
            const float* Xj = X + ((size_t)j * BATCH + b0) * DDIM;
            const int c4 = tid & 31;   // 32 float4 per row
            const int rb = tid >> 5;   // 0..7
#pragma unroll
            for (int p = 0; p < 16; ++p) {
                const int r = rb + p * 8;
                const float4 v = *(const float4*)(Xj + (size_t)r * DDIM + c4 * 4);
                float xs[4] = {v.x, v.y, v.z, v.w};
                Pack4 ph, pl;
#pragma unroll
                for (int k = 0; k < 4; ++k) {
                    unsigned short hb = f2bf_rne(xs[k]);
                    ph.h[k] = hb;
                    pl.h[k] = f2bf_rne(xs[k] - bf2f(hb));
                }
                *(uint2*)(XhL + r * SH + c4 * 4) = ph.u2;
                *(uint2*)(XlL + r * SH + c4 * 4) = pl.u2;
            }
        }
        // stage pre-split W[i][j] hi/lo into LDS (b128 copies)
        {
            const unsigned short* WhB = Wh + (size_t)(i * N_IN + j) * DDIM * DDIM;
            const unsigned short* WlB = Wl + (size_t)(i * N_IN + j) * DDIM * DDIM;
            const int c  = tid & 15;   // 16 uint4 per row
            const int rb = tid >> 4;   // 0..15
#pragma unroll
            for (int p = 0; p < 8; ++p) {
                const int r = rb + p * 16;
                *(uint4*)(WhL + r * SH + c * 8) = *(const uint4*)(WhB + r * DDIM + c * 8);
                *(uint4*)(WlL + r * SH + c * 8) = *(const uint4*)(WlB + r * DDIM + c * 8);
            }
        }
        __syncthreads();

#pragma unroll
        for (int kc = 0; kc < 4; ++kc) {
            const int kb = kc * 32;
            FragU Ah[2], Al[2];
#pragma unroll
            for (int mt = 0; mt < 2; ++mt) {
                // A 16x32 bf16: slots0-7 K=kb+8g.., slots8-15 K=kb+16+8g..
                const int off = (m0 + mt * 16 + lr) * SH + kb + g * 8;
                Ah[mt].u[0] = *(const uint4*)(XhL + off);
                Ah[mt].u[1] = *(const uint4*)(XhL + off + 16);
                Al[mt].u[0] = *(const uint4*)(XlL + off);
                Al[mt].u[1] = *(const uint4*)(XlL + off + 16);
            }
            FragU Bh[4], Bl[4];
#pragma unroll
            for (int nt = 0; nt < 4; ++nt) {
                // B 32x16 bf16 (from W[o][d]): slots0-7 K=kb+8g.., slots8-15 K=kb+16+8g..
                const int off = (n0 + nt * 16 + lr) * SH + kb + g * 8;
                Bh[nt].u[0] = *(const uint4*)(WhL + off);
                Bh[nt].u[1] = *(const uint4*)(WhL + off + 16);
                Bl[nt].u[0] = *(const uint4*)(WlL + off);
                Bl[nt].u[1] = *(const uint4*)(WlL + off + 16);
            }
#pragma unroll
            for (int mt = 0; mt < 2; ++mt)
#pragma unroll
                for (int nt = 0; nt < 4; ++nt) {
                    acc[mt][nt] = __builtin_amdgcn_wmma_f32_16x16x32_bf16(
                        false, Ah[mt].v, false, Bh[nt].v, (short)0, acc[mt][nt], false, false);
                    acc[mt][nt] = __builtin_amdgcn_wmma_f32_16x16x32_bf16(
                        false, Al[mt].v, false, Bh[nt].v, (short)0, acc[mt][nt], false, false);
                    acc[mt][nt] = __builtin_amdgcn_wmma_f32_16x16x32_bf16(
                        false, Ah[mt].v, false, Bl[nt].v, (short)0, acc[mt][nt], false, false);
                }
        }
    }

    // bias + store (D layout: N = lane&15, M = 8*(lane>>4) + reg)
    float bias[4];
#pragma unroll
    for (int nt = 0; nt < 4; ++nt) bias[nt] = biasSum[i * DDIM + n0 + nt * 16 + lr];
#pragma unroll
    for (int mt = 0; mt < 2; ++mt)
#pragma unroll
        for (int nt = 0; nt < 4; ++nt)
#pragma unroll
            for (int r = 0; r < 8; ++r) {
                const int M = m0 + mt * 16 + g * 8 + r;
                const int N = n0 + nt * 16 + lr;
                out[((size_t)i * BATCH + b0 + M) * DDIM + N] = acc[mt][nt][r] + bias[nt];
            }
}

extern "C" void kernel_launch(void* const* d_in, const int* in_sizes, int n_in,
                              void* d_out, int out_size, void* d_ws, size_t ws_size,
                              hipStream_t stream) {
    const float* X = (const float*)d_in[0];
    const float* W = (const float*)d_in[1];
    const float* b = (const float*)d_in[2];
    // d_in[3] (mask) ignored: band structure is a compile-time constant, as in the reference.
    float* out = (float*)d_out;

    unsigned short* Wh = (unsigned short*)d_ws;
    unsigned short* Wl = Wh + (size_t)N_OUT * N_IN * DDIM * DDIM;
    float* bs = (float*)(Wl + (size_t)N_OUT * N_IN * DDIM * DDIM);

    seq2d_prep_w<<<(N_OUT * N_IN * DDIM * DDIM) / (256 * 4), 256, 0, stream>>>(W, Wh, Wl);
    seq2d_prep_bias<<<(N_OUT * DDIM) / 256, 256, 0, stream>>>(b, bs);

    dim3 grid(BATCH / TILE_B, N_OUT);
    seq2d_main<<<grid, 256, 0, stream>>>(X, Wh, Wl, bs, out);
}
